// SelfModifyingTitansChunk_34041910788568
// MI455X (gfx1250) — matrix operand order repaired and checked by reading, not compiled, not run on hardware
//
#include <hip/hip_runtime.h>
#include <hip/hip_bf16.h>

typedef unsigned short u16;
typedef unsigned int   u32;

typedef __attribute__((ext_vector_type(16))) __bf16 v16bf;
typedef __attribute__((ext_vector_type(8)))  float  v8f;

union Frag {
    v16bf v;
    u32   u[8];
    uint4 q[2];
};

#define B_  4
#define S_  4096
#define DIM_ 768
#define H_  12
#define D_  64
#define C_  16
#define NC_ 256

__device__ __forceinline__ u16 f2bf(float f) {
    u32 u = __float_as_uint(f);
    u += 0x7fffu + ((u >> 16) & 1u);   // round-to-nearest-even
    return (u16)(u >> 16);
}
__device__ __forceinline__ u32 packbf(float lo, float hi) {
    return (u32)f2bf(lo) | ((u32)f2bf(hi) << 16);
}
__device__ __forceinline__ float sigmoidf_(float x) {
    return 1.f / (1.f + __expf(-x));
}

// ---------------------------------------------------------------------------
// 1) Input LayerNorm: x[B*S,768] f32 -> xn bf16
// ---------------------------------------------------------------------------
__global__ __launch_bounds__(256)
void ln_in_ker(const float* __restrict__ x, const float* __restrict__ g,
               const float* __restrict__ bb, u16* __restrict__ xnbf) {
    __shared__ float s1[256], s2[256];
    const int row = blockIdx.x;
    const int tid = threadIdx.x;
    const float* xr = x + (size_t)row * DIM_;
    float v[3], sum = 0.f, sq = 0.f;
    for (int i = 0; i < 3; ++i) {
        v[i] = xr[tid + 256 * i];
        sum += v[i]; sq += v[i] * v[i];
    }
    s1[tid] = sum; s2[tid] = sq;
    __syncthreads();
    for (int st = 128; st > 0; st >>= 1) {
        if (tid < st) { s1[tid] += s1[tid + st]; s2[tid] += s2[tid + st]; }
        __syncthreads();
    }
    float mu = s1[0] * (1.f / DIM_);
    float var = s2[0] * (1.f / DIM_) - mu * mu;
    float rs = rsqrtf(var + 1e-5f);
    u16* orow = xnbf + (size_t)row * DIM_;
    for (int i = 0; i < 3; ++i) {
        int c = tid + 256 * i;
        orow[c] = f2bf((v[i] - mu) * rs * g[c] + bb[c]);
    }
}

// ---------------------------------------------------------------------------
// 2) Weight transpose + f32->bf16: W[K,N] -> Wt[N,K] bf16
// ---------------------------------------------------------------------------
__global__ __launch_bounds__(256)
void transpose_bf16_ker(const float* __restrict__ W, u16* __restrict__ Wt,
                        int K, int N) {
    int idx = blockIdx.x * blockDim.x + threadIdx.x;
    if (idx >= K * N) return;
    int n = idx / K, k = idx % K;
    Wt[idx] = f2bf(W[(size_t)k * N + n]);
}

// ---------------------------------------------------------------------------
// 3) WMMA GEMM (M=16384, N=768, K=768 fixed), register-blocked,
//    double-buffered async-LDS B panel with a UNIFORM pipelined loop:
//    every stage prefetches the next slab (tail prefetch is a dummy) and
//    waits s_wait_asynccnt 1, so there is exactly one straight-line body
//    per buffer phase (unroll-by-2 makes the ping-pong index static).
// ---------------------------------------------------------------------------
__global__ __launch_bounds__(256)
void gemm_bf16_wmma(const u16* __restrict__ A, const u16* __restrict__ Bt,
                    float* __restrict__ Y,
                    const float* __restrict__ gate_lin,
                    const float* __restrict__ gate_b) {
    constexpr int N = DIM_;           // 768
    constexpr int K = DIM_;           // 768
    constexpr int KT = K / 32;        // 24 K-steps (even)

    __shared__ __align__(16) u16 bsh[2][64 * 32];   // 2 x (64 n-rows x 32 k)
    const int lane = threadIdx.x & 31;
    const int wave = threadIdx.x >> 5;
    const int nblk = N >> 6;                  // 12 N-blocks of 64
    const int m0 = (blockIdx.x / nblk) * 128 + wave * 16;
    const int n0 = (blockIdx.x % nblk) * 64;

    // cooperative B-slab mapping: thread -> one 16B async transfer
    const int nl = threadIdx.x >> 2;          // 0..63  (local n row)
    const int kp = (threadIdx.x & 3) * 8;     // 0,8,16,24 (bf16 in k-slice)
    const u32 lds_dst0 = (u32)(uintptr_t)(&bsh[0][nl * 32 + kp]);
    const u32 lds_dst1 = (u32)(uintptr_t)(&bsh[1][nl * 32 + kp]);
    const u16* gbase = Bt + (size_t)(n0 + nl) * K + kp;

    // A fragment base (ISA 16-bit A 16x32 layout): lane<16 -> kb=0, else kb=8
    const size_t arow = (size_t)(m0 + (lane & 15)) * K + ((lane >> 4) ? 8 : 0);
    const int khalf = (lane >> 4) ? 8 : 0;    // B fragment K base (table order, matches A)

    v8f acc[4] = {};

    // one pipeline stage: consume slab `rbase` for k-slice k0, prefetch
    // k-slice knext into the other buffer (wdst)
    auto stage = [&](const u16* rbase, u32 wdst, int k0, int knext) {
        const u16* gsrc = gbase + knext;
        asm volatile("global_load_async_to_lds_b128 %0, %1, off"
                     :: "v"(wdst), "v"(gsrc) : "memory");
        Frag a;
        a.q[0] = *reinterpret_cast<const uint4*>(A + arow + k0);
        a.q[1] = *reinterpret_cast<const uint4*>(A + arow + k0 + 16);
        asm volatile("s_wait_asynccnt 1" ::: "memory");
        __syncthreads();                          // writes to rbase visible
        Frag bfr[4];
        #pragma unroll
        for (int j = 0; j < 4; ++j) {
            const uint4* p = reinterpret_cast<const uint4*>(
                rbase + (j * 16 + (lane & 15)) * 32 + khalf);
            bfr[j].q[0] = p[0];
            bfr[j].q[1] = p[2];
        }
        #pragma unroll
        for (int j = 0; j < 4; ++j)
            acc[j] = __builtin_amdgcn_wmma_f32_16x16x32_bf16(
                false, a.v, false, bfr[j].v, (short)0, acc[j], false, false);
        __syncthreads();                          // reads done before overwrite
    };

    // prologue: stage slab 0 into buffer 0
    asm volatile("global_load_async_to_lds_b128 %0, %1, off"
                 :: "v"(lds_dst0), "v"(gbase) : "memory");

    #pragma unroll 1
    for (int kt = 0; kt < KT; kt += 2) {
        const int k0 = kt << 5;
        stage(&bsh[0][0], lds_dst1, k0, k0 + 32);                 // buffer 0
        stage(&bsh[1][0], lds_dst0, k0 + 32,
              (kt + 2 < KT) ? (k0 + 64) : 0 /* dummy tail */);    // buffer 1
    }
    asm volatile("s_wait_asynccnt 0" ::: "memory");  // drain dummy prefetch

    // C layout: VGPR i, lane<16 -> M=m0+i, lane>=16 -> M=m0+8+i
    const int srow = m0 + ((lane >> 4) ? 8 : 0);
    for (int j = 0; j < 4; ++j) {
        const int col = n0 + j * 16 + (lane & 15);
        for (int i = 0; i < 8; ++i) {
            float val = acc[j][i];
            if (gate_lin) {
                float gl = gate_lin[(size_t)(srow + i) * N + col] + gate_b[col];
                val *= sigmoidf_(gl);
            }
            Y[(size_t)(srow + i) * N + col] = val;
        }
    }
}

// ---------------------------------------------------------------------------
// 4) Per-head LayerNorm(64) + elu + 1, in place. One wave per row of 64.
// ---------------------------------------------------------------------------
__global__ __launch_bounds__(256)
void ln_head_ker(float* __restrict__ t, const float* __restrict__ g,
                 const float* __restrict__ bb, int nrows) {
    int r = blockIdx.x * (blockDim.x >> 5) + (threadIdx.x >> 5);
    int lane = threadIdx.x & 31;
    if (r >= nrows) return;
    int h = r % H_;
    size_t sr = (size_t)(r / H_);
    float* p = t + sr * DIM_ + h * D_;
    int c0 = lane * 2;
    float v0 = p[c0], v1 = p[c0 + 1];
    float sum = v0 + v1, sq = v0 * v0 + v1 * v1;
    for (int m = 16; m > 0; m >>= 1) {
        sum += __shfl_xor(sum, m, 32);
        sq  += __shfl_xor(sq,  m, 32);
    }
    float mu = sum * (1.f / D_);
    float var = sq * (1.f / D_) - mu * mu;
    float rs = rsqrtf(var + 1e-5f);
    float y0 = (v0 - mu) * rs * g[c0]     + bb[c0];
    float y1 = (v1 - mu) * rs * g[c0 + 1] + bb[c0 + 1];
    p[c0]     = (y0 > 0.f) ? y0 + 1.f : __expf(y0);   // elu(y)+1
    p[c0 + 1] = (y1 > 0.f) ? y1 + 1.f : __expf(y1);
}

// ---------------------------------------------------------------------------
// 5) Chunked scan: one wave per (b,h). Memory matrix [64x64] in LDS.
// ---------------------------------------------------------------------------
__global__ __launch_bounds__(32)
void chunk_scan_ker(const float* __restrict__ kf, const float* __restrict__ qf,
                    const float* __restrict__ vf, const float* __restrict__ mem0,
                    const float* __restrict__ lr_scale,
                    u16* __restrict__ attnbf, float* __restrict__ mem_out) {
    const int b = blockIdx.x / H_, h = blockIdx.x % H_;
    const int lane = threadIdx.x;

    __shared__ float mem[64][65];
    __shared__ float sc[16][17];
    __shared__ float khat[64];
    __shared__ float coef[64];

    for (int i = lane; i < 4096; i += 32)
        mem[i >> 6][i & 63] = mem0[(size_t)blockIdx.x * 4096 + i];
    const float lr = sigmoidf_(lr_scale[h]);
    __syncthreads();

    const size_t colbase = (size_t)h * D_;

    for (int nc = 0; nc < NC_; ++nc) {
        const size_t rb = (size_t)b * S_ + nc * C_;   // chunk base row in [B*S]

        // ---- q A-fragments (K = head dim d, two 32-wide slices)
        Frag qa[2];
        {
            const float* qrow = qf + (rb + (lane & 15)) * DIM_ + colbase;
            for (int kk = 0; kk < 2; ++kk) {
                int kb_ = kk * 32 + ((lane >> 4) ? 16 : 0);
                for (int v2 = 0; v2 < 4; ++v2) {
                    qa[kk].u[v2]     = packbf(qrow[kb_ + 2*v2],      qrow[kb_ + 2*v2 + 1]);
                    qa[kk].u[4 + v2] = packbf(qrow[kb_ + 8 + 2*v2], qrow[kb_ + 8 + 2*v2 + 1]);
                }
            }
        }
        // ---- scores = q @ k^T  (B[K=d, n=t] = k[t,d])
        v8f accS = {};
        {
            const float* krow = kf + (rb + (lane & 15)) * DIM_ + colbase;
            for (int kk = 0; kk < 2; ++kk) {
                Frag kb;
                int kc = kk * 32 + ((lane >> 4) ? 16 : 0);
                for (int v2 = 0; v2 < 8; ++v2)
                    kb.u[v2] = packbf(krow[kc + 2*v2], krow[kc + 2*v2 + 1]);
                accS = __builtin_amdgcn_wmma_f32_16x16x32_bf16(
                    false, qa[kk].v, false, kb.v, (short)0, accS, false, false);
            }
        }
        // ---- causal mask (t <= s), re-layout via LDS for A-fragment use
        {
            int srow = (lane >> 4) ? 8 : 0, t = lane & 15;
            for (int j = 0; j < 8; ++j) {
                int s = j + srow;
                sc[s][t] = (t <= s) ? accS[j] : 0.f;
            }
        }
        __syncthreads();
        Frag sa;   // scores as A-fragment, K (=t) zero-padded 16..31
        {
            int s = lane & 15;
            for (int v2 = 0; v2 < 8; ++v2) {
                sa.u[v2] = (lane < 16) ? packbf(sc[s][2*v2], sc[s][2*v2 + 1]) : 0u;
            }
        }
        // ---- out tiles: out[s,d] = scores@v + q@mem^T, 4 tiles of N=16
        for (int d0 = 0; d0 < D_; d0 += 16) {
            v8f acc = {};
            Frag vb;   // B[K=t, n=d] = v[t,d]; K>=16 zero (A padded too)
            if (lane < 16) {
                const float* vcol = vf + rb * DIM_ + colbase + d0 + lane;
                for (int v2 = 0; v2 < 8; ++v2)
                    vb.u[v2] = packbf(vcol[(2*v2) * DIM_], vcol[(2*v2 + 1) * DIM_]);
            } else {
                for (int v2 = 0; v2 < 8; ++v2) vb.u[v2] = 0u;
            }
            acc = __builtin_amdgcn_wmma_f32_16x16x32_bf16(
                false, sa.v, false, vb.v, (short)0, acc, false, false);
            for (int kk = 0; kk < 2; ++kk) {    // inter: B[K=k, n=d] = mem[d][k]
                Frag mb;
                int d = d0 + (lane & 15);
                int kc = kk * 32 + ((lane >> 4) ? 16 : 0);
                for (int v2 = 0; v2 < 8; ++v2)
                    mb.u[v2] = packbf(mem[d][kc + 2*v2], mem[d][kc + 2*v2 + 1]);
                acc = __builtin_amdgcn_wmma_f32_16x16x32_bf16(
                    false, qa[kk].v, false, mb.v, (short)0, acc, false, false);
            }
            int srow = (lane >> 4) ? 8 : 0, n = lane & 15;
            for (int j = 0; j < 8; ++j)
                attnbf[(rb + srow + j) * DIM_ + colbase + d0 + n] = f2bf(acc[j]);
        }
        __syncthreads();
        // ---- delta-rule memory update (chunk means, rank-1)
        {
            float kbv[2], vbv[2];
            for (int half = 0; half < 2; ++half) {
                int d = lane + half * 32;
                const float* kc0 = kf + rb * DIM_ + colbase + d;
                const float* vc0 = vf + rb * DIM_ + colbase + d;
                float sk = 0.f, sv = 0.f;
                for (int t = 0; t < C_; ++t) { sk += kc0[t * DIM_]; sv += vc0[t * DIM_]; }
                kbv[half] = sk * (1.f / C_);
                vbv[half] = sv * (1.f / C_);
            }
            float n2 = kbv[0]*kbv[0] + kbv[1]*kbv[1];
            for (int m = 16; m > 0; m >>= 1) n2 += __shfl_xor(n2, m, 32);
            float inv = 1.f / (sqrtf(n2) + 1e-6f);
            khat[lane]      = kbv[0] * inv;
            khat[lane + 32] = kbv[1] * inv;
            __syncthreads();
            for (int half = 0; half < 2; ++half) {
                int d = lane + half * 32;
                float pred = 0.f;
                for (int k2 = 0; k2 < D_; ++k2) pred += mem[d][k2] * khat[k2];
                float sur = pred - vbv[half];
                coef[d] = pred + lr * sur;        // mem -= coef * khat^T
            }
            __syncthreads();
            for (int half = 0; half < 2; ++half) {
                int d = lane + half * 32;
                float c = coef[d];
                for (int k2 = 0; k2 < D_; ++k2) mem[d][k2] -= c * khat[k2];
            }
            __syncthreads();
        }
    }
    for (int i = lane; i < 4096; i += 32)
        mem_out[(size_t)blockIdx.x * 4096 + i] = mem[i >> 6][i & 63];
}

// ---------------------------------------------------------------------------
extern "C" void kernel_launch(void* const* d_in, const int* in_sizes, int n_in,
                              void* d_out, int out_size, void* d_ws, size_t ws_size,
                              hipStream_t stream) {
    (void)in_sizes; (void)n_in; (void)out_size; (void)ws_size;
    const float* x        = (const float*)d_in[0];
    const float* memory0  = (const float*)d_in[1];
    const float* Wk       = (const float*)d_in[2];
    const float* Wv       = (const float*)d_in[3];
    const float* Wq       = (const float*)d_in[4];
    const float* Wo       = (const float*)d_in[5];
    const float* ln_in_g  = (const float*)d_in[6];
    const float* ln_in_b  = (const float*)d_in[7];
    const float* ln_k_g   = (const float*)d_in[8];
    const float* ln_k_b   = (const float*)d_in[9];
    const float* ln_q_g   = (const float*)d_in[10];
    const float* ln_q_b   = (const float*)d_in[11];
    const float* lr_scale = (const float*)d_in[12];
    // d_in[13] forget_scale: unused by reference
    const float* gate_W   = (const float*)d_in[14];
    const float* gate_b   = (const float*)d_in[15];

    const int M = B_ * S_;                 // 16384
    const size_t WN = (size_t)DIM_ * DIM_; // 589824

    auto al = [](size_t v) { return (v + 255) & ~(size_t)255; };
    char* ws = (char*)d_ws;
    size_t off = 0;
    u16* xnbf = (u16*)(ws + off); off = al(off + (size_t)M * DIM_ * 2);
    u16* wtK  = (u16*)(ws + off); off = al(off + WN * 2);
    u16* wtQ  = (u16*)(ws + off); off = al(off + WN * 2);
    u16* wtV  = (u16*)(ws + off); off = al(off + WN * 2);
    u16* wtG  = (u16*)(ws + off); off = al(off + WN * 2);
    u16* wtO  = (u16*)(ws + off); off = al(off + WN * 2);
    float* kf    = (float*)(ws + off); off = al(off + (size_t)M * DIM_ * 4);
    float* qf    = (float*)(ws + off); off = al(off + (size_t)M * DIM_ * 4);
    float* vf    = (float*)(ws + off); off = al(off + (size_t)M * DIM_ * 4);
    float* gatef = (float*)(ws + off); off = al(off + (size_t)M * DIM_ * 4);
    u16* attnbf  = (u16*)(ws + off);   off = al(off + (size_t)M * DIM_ * 2);

    // 1) input LN -> bf16
    ln_in_ker<<<M, 256, 0, stream>>>(x, ln_in_g, ln_in_b, xnbf);

    // 2) weight transpose+convert
    int tb = (int)((WN + 255) / 256);
    transpose_bf16_ker<<<tb, 256, 0, stream>>>(Wk,     wtK, DIM_, DIM_);
    transpose_bf16_ker<<<tb, 256, 0, stream>>>(Wq,     wtQ, DIM_, DIM_);
    transpose_bf16_ker<<<tb, 256, 0, stream>>>(Wv,     wtV, DIM_, DIM_);
    transpose_bf16_ker<<<tb, 256, 0, stream>>>(gate_W, wtG, DIM_, DIM_);
    transpose_bf16_ker<<<tb, 256, 0, stream>>>(Wo,     wtO, DIM_, DIM_);

    // 3) projection GEMMs (block = 128M x 64N, double-buffered async-LDS B)
    const int gblocks = (M / 128) * (DIM_ / 64);  // 128 * 12 = 1536
    gemm_bf16_wmma<<<gblocks, 256, 0, stream>>>(xnbf, wtK, kf,    nullptr, nullptr);
    gemm_bf16_wmma<<<gblocks, 256, 0, stream>>>(xnbf, wtQ, qf,    nullptr, nullptr);
    gemm_bf16_wmma<<<gblocks, 256, 0, stream>>>(xnbf, wtV, vf,    nullptr, nullptr);
    gemm_bf16_wmma<<<gblocks, 256, 0, stream>>>(xnbf, wtG, gatef, nullptr, nullptr);

    // 4) per-head LN + elu + 1 on k, q
    const int hr = M * H_;                        // 196608 rows of 64
    ln_head_ker<<<hr / 8, 256, 0, stream>>>(kf, ln_k_g, ln_k_b, hr);
    ln_head_ker<<<hr / 8, 256, 0, stream>>>(qf, ln_q_g, ln_q_b, hr);

    // 5) chunked scan -> attn (bf16) and memory_final (tail of d_out)
    float* out_f = (float*)d_out;
    float* mem_out = out_f + (size_t)M * DIM_;
    chunk_scan_ker<<<B_ * H_, 32, 0, stream>>>(kf, qf, vf, memory0, lr_scale,
                                               attnbf, mem_out);

    // 6) out @ Wo with fused sigmoid gate
    gemm_bf16_wmma<<<gblocks, 256, 0, stream>>>(attnbf, wtO, out_f, gatef, gate_b);
}
